// CRF_Loss_20220706029950
// MI455X (gfx1250) — compile-verified, matrix operand order checked by reading, not run on hardware
//
#include <hip/hip_runtime.h>

#define CRF_S     512
#define CRF_TAGS  64
#define CRF_NT    66
#define CRF_START 64
#define CRF_STOP  65

typedef __attribute__((ext_vector_type(16))) _Float16 v16h;
typedef __attribute__((ext_vector_type(8)))  float    v8f;

union V16HQ { v16h v; float4 q[2]; };

// Build a 16-half WMMA operand (A or B layout) from two contiguous 16B LDS runs.
// Per the CDNA5 16-bit A/B layout, lane (half h) holds K = {base+8h+0..7} and
// {base+16+8h+0..7} packed pairwise into 8 VGPRs.
__device__ __forceinline__ v16h ldsOp16(const _Float16* base) {
  V16HQ u;
  u.q[0] = *(const float4*)(base);
  u.q[1] = *(const float4*)(base + 16);
  return u.v;
}

__launch_bounds__(32)
__global__ void crf_forward_wmma(const float* __restrict__ em,
                                 const int*   __restrict__ tags,
                                 const float* __restrict__ T,
                                 float*       __restrict__ logz)
{
  // LDS: exp(Tsub)^T as f16 [N][K], alphaExp f16 [M][K], em tile f32 [M][N]
  __shared__ __align__(32) _Float16 expTt[CRF_TAGS * CRF_TAGS];
  __shared__ __align__(32) _Float16 aT[16 * CRF_TAGS];
  __shared__ __align__(16) float    emT[16 * CRF_TAGS];
  __shared__ __align__(16) int      tagT[16];

  const int tid  = threadIdx.x;
  const int half = tid >> 4;
  const int mrow = tid & 15;
  const int bb   = blockIdx.x * 16;

  // exp(Tsub) transposed into LDS: expTt[n*64+k] = exp(T[k][n])
  for (int idx = tid; idx < CRF_TAGS * CRF_TAGS; idx += 32) {
    int n = idx >> 6, k = idx & 63;
    expTt[idx] = (_Float16)__expf(T[k * CRF_NT + n]);
  }

  // maxTend over Tend[j] = T[j][STOP]  (wave-wide max reduction)
  float mt = fmaxf(T[tid * CRF_NT + CRF_STOP], T[(tid + 32) * CRF_NT + CRF_STOP]);
  #pragma unroll
  for (int s = 16; s > 0; s >>= 1) mt = fmaxf(mt, __shfl_xor(mt, s, 32));
  const float maxTend = mt;

  float eTend[4];
  #pragma unroll
  for (int nt = 0; nt < 4; ++nt)
    eTend[nt] = __expf(T[(nt * 16 + mrow) * CRF_NT + CRF_STOP] - maxTend);

  __syncthreads();

  // Resident B operands (expT), 4 N-tiles x 2 K-chunks
  v16h Bv[4][2];
  #pragma unroll
  for (int nt = 0; nt < 4; ++nt)
    #pragma unroll
    for (int c = 0; c < 2; ++c)
      Bv[nt][c] = ldsOp16(expTt + (nt * 16 + mrow) * 64 + c * 32 + 8 * half);

  // Stage emissions tile for t = 0 (coalesced float4)
  for (int i = tid; i < 16 * 16; i += 32) {
    int row = i >> 4, c4 = i & 15;
    ((float4*)emT)[i] =
        *(const float4*)(em + ((size_t)(bb + row) * CRF_S) * CRF_TAGS + c4 * 4);
  }
  __syncthreads();

  float alpha[4][8];  // alphaExp in C/D layout: [N-tile][M-vgpr]
  float Crow[8];      // per-row running log-scale (uniform across the 16-lane half)

  // init: log_alpha0 = em[:,0,:] + Tstart ; normalize per row
  {
    float v[4][8];
    #pragma unroll
    for (int nt = 0; nt < 4; ++nt) {
      float ts = T[CRF_START * CRF_NT + nt * 16 + mrow];
      #pragma unroll
      for (int k = 0; k < 8; ++k)
        v[nt][k] = emT[(k + 8 * half) * 64 + nt * 16 + mrow] + ts;
    }
    #pragma unroll
    for (int k = 0; k < 8; ++k) {
      float m0 = fmaxf(fmaxf(v[0][k], v[1][k]), fmaxf(v[2][k], v[3][k]));
      #pragma unroll
      for (int s = 8; s > 0; s >>= 1) m0 = fmaxf(m0, __shfl_xor(m0, s, 32));
      Crow[k] = m0;
      #pragma unroll
      for (int nt = 0; nt < 4; ++nt) {
        alpha[nt][k] = __expf(v[nt][k] - m0);
        aT[(k + 8 * half) * 64 + nt * 16 + mrow] = (_Float16)alpha[nt][k];
      }
    }
  }

  for (int t = 1; t < CRF_S; ++t) {
    __syncthreads();  // prev-iter aT writes / emT reads complete
    for (int i = tid; i < 16 * 16; i += 32) {
      int row = i >> 4, c4 = i & 15;
      ((float4*)emT)[i] =
          *(const float4*)(em + ((size_t)(bb + row) * CRF_S + t) * CRF_TAGS + c4 * 4);
    }
    if (tid < 16) {
      tagT[tid] = tags[(size_t)(bb + tid) * CRF_S + t];
      if (t + 1 < CRF_S)
        __builtin_prefetch(em + ((size_t)(bb + tid) * CRF_S + t + 1) * CRF_TAGS, 0, 0);
    }
    __syncthreads();

    // A operands: 16 batch rows x K=64 of alphaExp (f16), two b128 LDS loads each
    v16h A0 = ldsOp16(aT + mrow * 64 +  0 + 8 * half);
    v16h A1 = ldsOp16(aT + mrow * 64 + 32 + 8 * half);

    // dot = alphaExp @ expT  via f16 WMMA, fp32 accumulate (K=64 => 2 chained WMMAs)
    float p[4][8];
    #pragma unroll
    for (int nt = 0; nt < 4; ++nt) {
      v8f d = {};
      d = __builtin_amdgcn_wmma_f32_16x16x32_f16(false, A0, false, Bv[nt][0],
                                                 (short)0, d, false, false);
      d = __builtin_amdgcn_wmma_f32_16x16x32_f16(false, A1, false, Bv[nt][1],
                                                 (short)0, d, false, false);
      #pragma unroll
      for (int k = 0; k < 8; ++k)
        p[nt][k] = __expf(emT[(k + 8 * half) * 64 + nt * 16 + mrow]) * d[k];
    }

    int tg[8];
    *(int4*)&tg[0] = ((const int4*)tagT)[half * 2 + 0];
    *(int4*)&tg[4] = ((const int4*)tagT)[half * 2 + 1];

    #pragma unroll
    for (int k = 0; k < 8; ++k) {
      float m0 = fmaxf(fmaxf(p[0][k], p[1][k]), fmaxf(p[2][k], p[3][k]));
      #pragma unroll
      for (int s = 8; s > 0; s >>= 1) m0 = fmaxf(m0, __shfl_xor(m0, s, 32));
      const bool mk = (tg[k] != 0);
      const float rinv = 1.0f / m0;
      Crow[k] += maxTend + (mk ? __logf(m0) : 0.0f);
      #pragma unroll
      for (int nt = 0; nt < 4; ++nt) {
        float na = mk ? p[nt][k] * rinv : alpha[nt][k];
        na *= eTend[nt];
        alpha[nt][k] = na;
        aT[(k + 8 * half) * 64 + nt * 16 + mrow] = (_Float16)na;
      }
    }
  }

  // log_z = Crow + log(sum_j alphaExp)
  #pragma unroll
  for (int k = 0; k < 8; ++k) {
    float s0 = (alpha[0][k] + alpha[1][k]) + (alpha[2][k] + alpha[3][k]);
    #pragma unroll
    for (int s = 8; s > 0; s >>= 1) s0 += __shfl_xor(s0, s, 32);
    if (mrow == 0)
      logz[bb + k + 8 * half] = Crow[k] + __logf(s0);
  }
}

// Gold-path score + final combine: out[b] = llh[b] - log_z[b]
__global__ void crf_gold(const float* __restrict__ em, const int* __restrict__ tags,
                         const float* __restrict__ T, const float* __restrict__ logz,
                         float* __restrict__ out, int B)
{
  int b = blockIdx.x * blockDim.x + threadIdx.x;
  if (b >= B) return;
  const int*   tr = tags + (size_t)b * CRF_S;
  const float* er = em + (size_t)b * CRF_S * CRF_TAGS;
  int t0 = tr[0];
  float msum = (t0 != 0) ? 1.0f : 0.0f;
  float llh = T[CRF_START * CRF_NT + t0] + er[t0] * msum;
  int tprev = t0;
  for (int t = 1; t < CRF_S; ++t) {
    int tg = tr[t];
    float m = (tg != 0) ? 1.0f : 0.0f;
    llh += (er[(size_t)t * CRF_TAGS + tg] + T[tprev * CRF_NT + tg]) * m;
    msum += m;
    tprev = tg;
  }
  int last = (int)(msum - 1.0f);
  if (last < 0) last = 0;
  if (last > CRF_S - 1) last = CRF_S - 1;
  int lastTag = tr[last];
  llh += T[lastTag * CRF_NT + CRF_STOP];
  out[b] = llh - logz[b];
}

extern "C" void kernel_launch(void* const* d_in, const int* in_sizes, int n_in,
                              void* d_out, int out_size, void* d_ws, size_t ws_size,
                              hipStream_t stream) {
  const float* em   = (const float*)d_in[0];
  const int*   tags = (const int*)d_in[1];
  const float* T    = (const float*)d_in[2];
  float* out  = (float*)d_out;
  float* logz = (float*)d_ws;

  const int B = in_sizes[1] / CRF_S;   // tags is B x S
  crf_forward_wmma<<<B / 16, 32, 0, stream>>>(em, tags, T, logz);
  crf_gold<<<(B + 255) / 256, 256, 0, stream>>>(em, tags, T, logz, out, B);
}
